// SOMP_90400471646853
// MI455X (gfx1250) — compile-verified, matrix operand order checked by reading, not run on hardware
//
#include <hip/hip_runtime.h>
#include <hip/hip_bf16.h>
#include <math.h>

// ---------------------------------------------------------------------------
// SOMP on MI455X (gfx1250): fused WMMA-bf16 scoring GEMM + f32 small algebra.
// ---------------------------------------------------------------------------

typedef __attribute__((ext_vector_type(16))) __bf16 v16bf;
typedef __attribute__((ext_vector_type(8)))  __bf16 v8bf;
typedef __attribute__((ext_vector_type(8)))  float  v8f;

#define EPSV 1e-8f
#define KITER 16   // k is a python scalar (=16) living in device mem; host loop count must be static

// ---------------- generic small kernels ----------------

__global__ void zero_f32_kernel(float* __restrict__ p, long n) {
    long i = (long)blockIdx.x * blockDim.x + threadIdx.x;
    if (i < n) p[i] = 0.0f;
}

__global__ void zero_i32_kernel(int* __restrict__ p, long n) {
    long i = (long)blockIdx.x * blockDim.x + threadIdx.x;
    if (i < n) p[i] = 0;
}

__global__ void cvt_bf16_kernel(const float* __restrict__ src, __bf16* __restrict__ dst, long n) {
    long i = (long)blockIdx.x * blockDim.x + threadIdx.x;
    if (i < n) dst[i] = (__bf16)src[i];
}

// Xc = X - mean; residual(out) = Xc
__global__ void center_kernel(const float* __restrict__ X, const float* __restrict__ mean,
                              float* __restrict__ Xc, float* __restrict__ resid,
                              long total, int D) {
    long i = (long)blockIdx.x * blockDim.x + threadIdx.x;
    if (i < total) {
        float v = X[i] - mean[(int)(i % D)];
        Xc[i] = v;
        resid[i] = v;
    }
}

// column sums / sums-of-squares via per-block partial + atomic (grid: D/256 x rowchunks)
__global__ __launch_bounds__(256)
void col_accum_kernel(const float* __restrict__ A, float* __restrict__ csum,
                      float* __restrict__ csq, int N, int D) {
    int c  = blockIdx.x * 256 + threadIdx.x;
    int r0 = blockIdx.y * 256;
    int r1 = r0 + 256; if (r1 > N) r1 = N;
    float s = 0.f, q = 0.f;
    for (int r = r0; r < r1; ++r) {
        float v = A[(size_t)r * D + c];
        s += v; q += v * v;
    }
    atomicAdd(&csum[c], s);
    atomicAdd(&csq[c],  q);
}

// mean[d] = csum/N ; stdsum[0] = sum_d var(X[:,d], ddof=1)
__global__ __launch_bounds__(256)
void finalize_mean_kernel(const float* __restrict__ csum, const float* __restrict__ csq,
                          float* __restrict__ mean, float* __restrict__ stdsum, int N, int D) {
    __shared__ float sh[256];
    float acc = 0.f;
    float fN = (float)N;
    for (int d = threadIdx.x; d < D; d += 256) {
        float s = csum[d], q = csq[d];
        mean[d] = s / fN;
        acc += (q - s * s / fN) / (fN - 1.0f);
    }
    sh[threadIdx.x] = acc;
    __syncthreads();
    for (int off = 128; off; off >>= 1) {
        if (threadIdx.x < off) sh[threadIdx.x] += sh[threadIdx.x + off];
        __syncthreads();
    }
    if (threadIdx.x == 0) stdsum[0] = sh[0];
}

// out[row] = ||A[row,:]|| (block per row; D = row length)
__global__ __launch_bounds__(256)
void row_norm_kernel(const float* __restrict__ A, float* __restrict__ out, int D) {
    __shared__ float sh[256];
    const float* r = A + (size_t)blockIdx.x * D;
    float s = 0.f;
    for (int d = threadIdx.x; d < D; d += 256) { float v = r[d]; s += v * v; }
    sh[threadIdx.x] = s;
    __syncthreads();
    for (int off = 128; off; off >>= 1) {
        if (threadIdx.x < off) sh[threadIdx.x] += sh[threadIdx.x + off];
        __syncthreads();
    }
    if (threadIdx.x == 0) out[blockIdx.x] = sqrtf(sh[0]);
}

// ---------------- fused scoring GEMM (WMMA bf16) ----------------
// scores[m] += sum_n | residual[n,:] . dict[m,:] |
// Block tile: 64 rows(X) x 128 cols(dict); 8 waves, each a 32x32 macro-tile
// of 2x2 v_wmma_f32_16x16x32_bf16 fragments. K loop over D in steps of 32.

__device__ inline v16bf ld_frag_bf16(const __bf16* __restrict__ p) {
    v8bf lo = *reinterpret_cast<const v8bf*>(p);        // K+0..7   (or +8..15 for hi lanes)
    v8bf hi = *reinterpret_cast<const v8bf*>(p + 16);   // K+16..23 (or +24..31)
    return __builtin_shufflevector(lo, hi, 0,1,2,3,4,5,6,7,8,9,10,11,12,13,14,15);
}

__global__ __launch_bounds__(256)
void score_gemm_kernel(const __bf16* __restrict__ Rb,   // [N,D] residual (bf16)
                       const __bf16* __restrict__ Db,   // [M,D] dictionary (bf16)
                       float* __restrict__ scores, int D) {
    const int lane = threadIdx.x & 31;
    const int wave = threadIdx.x >> 5;
    const int wr = wave >> 2;            // 0..1
    const int wc = wave & 3;             // 0..3
    const int rowBase = blockIdx.x * 64  + wr * 32;   // X rows
    const int colBase = blockIdx.y * 128 + wc * 32;   // dict atoms
    const int frow  = lane & 15;
    const int khalf = (lane >> 4) * 8;   // K sub-offset per lane half

    v8f acc00 = {}; v8f acc01 = {}; v8f acc10 = {}; v8f acc11 = {};

    const __bf16* a0p = Rb + (size_t)(rowBase      + frow) * D;
    const __bf16* a1p = Rb + (size_t)(rowBase + 16 + frow) * D;
    const __bf16* b0p = Db + (size_t)(colBase      + frow) * D;
    const __bf16* b1p = Db + (size_t)(colBase + 16 + frow) * D;

    for (int k0 = 0; k0 < D; k0 += 32) {
        const int ka = k0 + khalf;
        v16bf a0 = ld_frag_bf16(a0p + ka);
        v16bf a1 = ld_frag_bf16(a1p + ka);
        v16bf b0 = ld_frag_bf16(b0p + ka);
        v16bf b1 = ld_frag_bf16(b1p + ka);
        acc00 = __builtin_amdgcn_wmma_f32_16x16x32_bf16(false, a0, false, b0, (short)0, acc00, false, false);
        acc01 = __builtin_amdgcn_wmma_f32_16x16x32_bf16(false, a0, false, b1, (short)0, acc01, false, false);
        acc10 = __builtin_amdgcn_wmma_f32_16x16x32_bf16(false, a1, false, b0, (short)0, acc10, false, false);
        acc11 = __builtin_amdgcn_wmma_f32_16x16x32_bf16(false, a1, false, b1, (short)0, acc11, false, false);
    }

    // abs-reduce over the 16-row dimension of each fragment (C layout: lane%16 = col,
    // VGPR r = row r / r+8 depending on lane half) -> per-column partial.
    float p0 = 0.f, p1 = 0.f;
#pragma unroll
    for (int r = 0; r < 8; ++r) {
        p0 += fabsf(acc00[r]) + fabsf(acc10[r]);
        p1 += fabsf(acc01[r]) + fabsf(acc11[r]);
    }
    p0 += __shfl_xor(p0, 16, 32);   // fold row halves (same column, lanes l and l+16)
    p1 += __shfl_xor(p1, 16, 32);
    if (lane < 16) {
        atomicAdd(scores + colBase      + lane, p0);
        atomicAdd(scores + colBase + 16 + lane, p1);
    }
}

// ---------------- argmax over masked scores ----------------

__global__ __launch_bounds__(1024)
void argmax_mask_kernel(const float* __restrict__ scores, int* __restrict__ mask,
                        int* __restrict__ chosen, int M, int iter) {
    __shared__ float sv[1024];
    __shared__ int   si[1024];
    float best = -1.0f; int bi = 0;
    for (int m = threadIdx.x; m < M; m += 1024) {
        if (!mask[m]) {
            float s = scores[m];
            if (s > best) { best = s; bi = m; }
        }
    }
    sv[threadIdx.x] = best; si[threadIdx.x] = bi;
    __syncthreads();
    for (int off = 512; off; off >>= 1) {
        if (threadIdx.x < off) {
            float ov = sv[threadIdx.x + off]; int oi = si[threadIdx.x + off];
            if (ov > sv[threadIdx.x] || (ov == sv[threadIdx.x] && oi < si[threadIdx.x])) {
                sv[threadIdx.x] = ov; si[threadIdx.x] = oi;
            }
        }
        __syncthreads();
    }
    if (threadIdx.x == 0) { chosen[iter] = si[0]; mask[si[0]] = 1; }
}

// ---------------- incremental least squares ----------------

// AXc[iter, n] = dict[chosen[iter]] . Xc[n]   (wave per row)
__global__ __launch_bounds__(256)
void corr_row_kernel(const float* __restrict__ dict, const float* __restrict__ Xc,
                     const int* __restrict__ chosen, float* __restrict__ AXc,
                     int iter, int N, int D) {
    int wave = threadIdx.x >> 5, lane = threadIdx.x & 31;
    int n = blockIdx.x * 8 + wave;
    const float* a = dict + (size_t)chosen[iter] * D;
    const float* x = Xc   + (size_t)n * D;
    float s = 0.f;
    for (int d = lane; d < D; d += 32) s += a[d] * x[d];
#pragma unroll
    for (int off = 16; off; off >>= 1) s += __shfl_xor(s, off, 32);
    if (lane == 0) AXc[(size_t)iter * N + n] = s;
}

// G[iter, j] = G[j, iter] = atoms[iter] . atoms[j], j <= iter  (wave j)
__global__ __launch_bounds__(512)
void gram_row_kernel(const float* __restrict__ dict, const int* __restrict__ chosen,
                     float* __restrict__ G, int iter, int D) {
    int wave = threadIdx.x >> 5, lane = threadIdx.x & 31;
    if (wave > iter) return;
    const float* ai = dict + (size_t)chosen[iter] * D;
    const float* aj = dict + (size_t)chosen[wave] * D;
    float s = 0.f;
    for (int d = lane; d < D; d += 32) s += ai[d] * aj[d];
#pragma unroll
    for (int off = 16; off; off >>= 1) s += __shfl_xor(s, off, 32);
    if (lane == 0) { G[iter * 16 + wave] = s; G[wave * 16 + iter] = s; }
}

// 16x16 padded Cholesky (single thread; s <= 16; rows >= s become identity)
__global__ void cholesky16_kernel(const float* __restrict__ G, float* __restrict__ L, int s) {
    if (threadIdx.x != 0 || blockIdx.x != 0) return;
    float Mm[16][16];
    for (int j = 0; j < 16; ++j)
        for (int p = 0; p < 16; ++p) Mm[j][p] = 0.f;
    for (int j = 0; j < s; ++j)
        for (int p = 0; p <= j; ++p) Mm[j][p] = G[j * 16 + p];
    for (int j = 0; j < s; ++j) {
        float d = Mm[j][j];
        for (int p = 0; p < j; ++p) d -= Mm[j][p] * Mm[j][p];
        d = sqrtf(fmaxf(d, 1e-20f));
        Mm[j][j] = d;
        for (int r = j + 1; r < s; ++r) {
            float t = Mm[r][j];
            for (int p = 0; p < j; ++p) t -= Mm[r][p] * Mm[j][p];
            Mm[r][j] = t / d;
        }
    }
    for (int j = s; j < 16; ++j) Mm[j][j] = 1.0f;
    for (int j = 0; j < 16; ++j)
        for (int p = 0; p < 16; ++p) L[j * 16 + p] = Mm[j][p];
}

// W[:, n] = G^-1 AXc[:, n] via padded Cholesky; fully unrolled (registers)
__global__ __launch_bounds__(256)
void solve16_kernel(const float* __restrict__ L, const float* __restrict__ AXc,
                    float* __restrict__ W, int s, int N) {
    __shared__ float Ls[256];
    Ls[threadIdx.x] = L[threadIdx.x];
    __syncthreads();
    int n = blockIdx.x * 256 + threadIdx.x;
    float y[16];
#pragma unroll
    for (int j = 0; j < 16; ++j) y[j] = (j < s) ? AXc[(size_t)j * N + n] : 0.f;
#pragma unroll
    for (int j = 0; j < 16; ++j) {          // forward: L y' = y
        float t = y[j];
#pragma unroll
        for (int p = 0; p < 16; ++p) if (p < j) t -= Ls[j * 16 + p] * y[p];
        y[j] = t / Ls[j * 16 + j];
    }
#pragma unroll
    for (int j = 15; j >= 0; --j) {         // backward: L^T x = y'
        float t = y[j];
#pragma unroll
        for (int p = 0; p < 16; ++p) if (p > j) t -= Ls[p * 16 + j] * y[p];
        y[j] = t / Ls[j * 16 + j];
    }
#pragma unroll
    for (int j = 0; j < 16; ++j) if (j < s) W[(size_t)j * N + n] = y[j];
}

// recon[n,d] = sum_j atoms[j,d] * W[j,n];  resid = Xc - recon
__global__ __launch_bounds__(256)
void recon_resid_kernel(const float* __restrict__ dict, const float* __restrict__ Xc,
                        const float* __restrict__ W, const int* __restrict__ chosen,
                        float* __restrict__ recon, float* __restrict__ resid,
                        int s, int N, int D) {
    __shared__ float wv[16];
    __shared__ int   ai[16];
    int n = blockIdx.x;
    if (threadIdx.x < 16) {
        int ok = threadIdx.x < s;
        wv[threadIdx.x] = ok ? W[(size_t)threadIdx.x * N + n] : 0.f;
        ai[threadIdx.x] = chosen[ok ? threadIdx.x : 0];
    }
    __syncthreads();
    int d = blockIdx.y * 256 + threadIdx.x;
    float r = 0.f;
    for (int j = 0; j < s; ++j) r += dict[(size_t)ai[j] * D + d] * wv[j];
    size_t o = (size_t)n * D + d;
    recon[o] = r;
    resid[o] = Xc[o] - r;
}

// ---------------- per-iteration statistics ----------------

// per-row: cosine contribution and residual^2 sum (block per row)
__global__ __launch_bounds__(256)
void row_stats_kernel(const float* __restrict__ X, const float* __restrict__ recon,
                      const float* __restrict__ resid, const float* __restrict__ mean,
                      const float* __restrict__ xnorm, float* __restrict__ iterscal, int D) {
    __shared__ float s0[256], s1[256], s2[256];
    int n = blockIdx.x;
    const float* xr = X     + (size_t)n * D;
    const float* rr = recon + (size_t)n * D;
    const float* er = resid + (size_t)n * D;
    float dot = 0.f, rq = 0.f, resq = 0.f;
    for (int d = threadIdx.x; d < D; d += 256) {
        float rf = rr[d] + mean[d];
        float xv = xr[d];
        float ev = er[d];
        dot += xv * rf; rq += rf * rf; resq += ev * ev;
    }
    s0[threadIdx.x] = dot; s1[threadIdx.x] = rq; s2[threadIdx.x] = resq;
    __syncthreads();
    for (int off = 128; off; off >>= 1) {
        if (threadIdx.x < off) {
            s0[threadIdx.x] += s0[threadIdx.x + off];
            s1[threadIdx.x] += s1[threadIdx.x + off];
            s2[threadIdx.x] += s2[threadIdx.x + off];
        }
        __syncthreads();
    }
    if (threadIdx.x == 0) {
        float nx = fmaxf(xnorm[n], EPSV);
        float nr = fmaxf(sqrtf(s1[0]), EPSV);
        atomicAdd(&iterscal[0], s0[0] / (nx * nr));  // cosine sum
        atomicAdd(&iterscal[1], s2[0]);              // residual^2 sum
    }
}

__global__ __launch_bounds__(256)
void finalize_stats_kernel(const float* __restrict__ csum, const float* __restrict__ csq,
                           const float* __restrict__ iterscal, const float* __restrict__ stdsum,
                           float* __restrict__ out_evr, float* __restrict__ out_l2,
                           float* __restrict__ out_cos, int iter, int N, int D) {
    __shared__ float sh[256];
    float fN = (float)N;
    float acc = 0.f;
    for (int d = threadIdx.x; d < D; d += 256) {
        float s = csum[d], q = csq[d];
        acc += (q - s * s / fN) / (fN - 1.0f);
    }
    sh[threadIdx.x] = acc;
    __syncthreads();
    for (int off = 128; off; off >>= 1) {
        if (threadIdx.x < off) sh[threadIdx.x] += sh[threadIdx.x + off];
        __syncthreads();
    }
    if (threadIdx.x == 0) {
        out_evr[iter] = sh[0] / stdsum[0];
        out_l2[iter]  = iterscal[1] / (fN * (float)D);
        out_cos[iter] = iterscal[0] / fN;
    }
}

// ---------------- final outputs ----------------

__global__ void emit_chosen_kernel(const int* __restrict__ chosen, float* __restrict__ out, int k) {
    if (threadIdx.x < k) out[threadIdx.x] = (float)chosen[threadIdx.x];
}

__global__ void transpose_W_kernel(const float* __restrict__ W, float* __restrict__ out,
                                   int N, int k) {
    long idx = (long)blockIdx.x * blockDim.x + threadIdx.x;
    if (idx < (long)N * k) {
        int n = (int)(idx / k), j = (int)(idx % k);
        out[idx] = W[(size_t)j * N + n];
    }
}

__global__ void add_mean_kernel(float* __restrict__ recon, const float* __restrict__ mean,
                                long total, int D) {
    long i = (long)blockIdx.x * blockDim.x + threadIdx.x;
    if (i < total) recon[i] += mean[(int)(i % D)];
}

// ---------------------------------------------------------------------------
// host orchestration
// ---------------------------------------------------------------------------

extern "C" void kernel_launch(void* const* d_in, const int* in_sizes, int n_in,
                              void* d_out, int out_size, void* d_ws, size_t ws_size,
                              hipStream_t stream) {
    const float* X    = (const float*)d_in[0];
    const float* dict = (const float*)d_in[1];
    // d_in[2] (descriptors) unused by the reference math; d_in[3] is k (== KITER).

    const int D = 1024;
    const int N = in_sizes[0] / D;   // 16384
    const int M = in_sizes[1] / D;   // 4096
    const long ND = (long)N * D;
    const long MD = (long)M * D;

    // ---- workspace bump allocator ----
    char* w = (char*)d_ws;
    auto alloc = [&](size_t bytes) -> void* {
        void* p = (void*)w;
        w += (bytes + 255) & ~(size_t)255;
        return p;
    };
    float* mean     = (float*)alloc((size_t)D * 4);
    float* xnorm    = (float*)alloc((size_t)N * 4);
    float* scores   = (float*)alloc((size_t)M * 4);   // |-- per-iteration zero span
    float* colsum   = (float*)alloc((size_t)D * 4);   // |
    float* colsq    = (float*)alloc((size_t)D * 4);   // |
    float* iterscal = (float*)alloc(8 * 4);           // |-- [0]=cos_sum [1]=l2_sum
    float* stdsum   = (float*)alloc(4 * 4);
    float* G        = (float*)alloc(256 * 4);
    float* L        = (float*)alloc(256 * 4);
    float* AXc      = (float*)alloc((size_t)16 * N * 4);
    float* W        = (float*)alloc((size_t)16 * N * 4);
    float* Xc       = (float*)alloc((size_t)ND * 4);
    int*   mask     = (int*)  alloc((size_t)M * 4);
    int*   chosen   = (int*)  alloc(16 * 4);
    __bf16* dictb   = (__bf16*)alloc((size_t)MD * 2);
    __bf16* residb  = (__bf16*)alloc((size_t)ND * 2);
    if ((size_t)(w - (char*)d_ws) > ws_size) return;  // workspace too small

    const long iterZeroSpan = ((char*)stdsum - (char*)scores) / 4; // scores+colsum+colsq+iterscal
    const long setupZeroSpan = ((char*)G - (char*)scores) / 4;     // + stdsum

    // ---- d_out layout (flat concat, reference return order) ----
    float* out       = (float*)d_out;
    float* o_recon   = out;                    // [N, D]
    float* o_resid   = o_recon + ND;           // [N, D]  (doubles as live residual buffer)
    float* o_chosen  = o_resid + ND;           // [16]
    float* o_weights = o_chosen + KITER;       // [16]
    float* o_WT      = o_weights + KITER;      // [N, 16]
    float* o_evr     = o_WT + (long)N * KITER; // [16]
    float* o_l2      = o_evr + KITER;          // [16]
    float* o_cos     = o_l2 + KITER;           // [16]

    const dim3 b256(256);

    // ---- setup ----
    zero_f32_kernel<<<(int)((setupZeroSpan + 255) / 256), b256, 0, stream>>>(scores, setupZeroSpan);
    zero_i32_kernel<<<(M + 255) / 256, b256, 0, stream>>>(mask, M);
    zero_i32_kernel<<<1, 32, 0, stream>>>(chosen, 16);
    col_accum_kernel<<<dim3(D / 256, (N + 255) / 256), b256, 0, stream>>>(X, colsum, colsq, N, D);
    finalize_mean_kernel<<<1, b256, 0, stream>>>(colsum, colsq, mean, stdsum, N, D);
    row_norm_kernel<<<N, b256, 0, stream>>>(X, xnorm, D);
    center_kernel<<<(int)((ND + 255) / 256), b256, 0, stream>>>(X, mean, Xc, o_resid, ND, D);
    cvt_bf16_kernel<<<(int)((MD + 255) / 256), b256, 0, stream>>>(dict, dictb, MD);

    // ---- k greedy iterations ----
    for (int it = 0; it < KITER; ++it) {
        const int s = it + 1;
        cvt_bf16_kernel<<<(int)((ND + 255) / 256), b256, 0, stream>>>(o_resid, residb, ND);
        zero_f32_kernel<<<(int)((iterZeroSpan + 255) / 256), b256, 0, stream>>>(scores, iterZeroSpan);
        score_gemm_kernel<<<dim3(N / 64, M / 128), b256, 0, stream>>>(residb, dictb, scores, D);
        argmax_mask_kernel<<<1, 1024, 0, stream>>>(scores, mask, chosen, M, it);
        corr_row_kernel<<<N / 8, b256, 0, stream>>>(dict, Xc, chosen, AXc, it, N, D);
        gram_row_kernel<<<1, 512, 0, stream>>>(dict, chosen, G, it, D);
        cholesky16_kernel<<<1, 32, 0, stream>>>(G, L, s);
        solve16_kernel<<<N / 256, b256, 0, stream>>>(L, AXc, W, s, N);
        recon_resid_kernel<<<dim3(N, D / 256), b256, 0, stream>>>(dict, Xc, W, chosen,
                                                                  o_recon, o_resid, s, N, D);
        col_accum_kernel<<<dim3(D / 256, (N + 255) / 256), b256, 0, stream>>>(o_recon, colsum, colsq, N, D);
        row_stats_kernel<<<N, b256, 0, stream>>>(X, o_recon, o_resid, mean, xnorm, iterscal, D);
        finalize_stats_kernel<<<1, b256, 0, stream>>>(colsum, colsq, iterscal, stdsum,
                                                      o_evr, o_l2, o_cos, it, N, D);
    }

    // ---- final outputs ----
    row_norm_kernel<<<KITER, b256, 0, stream>>>(W, o_weights, N);                 // ||W[j,:]||
    emit_chosen_kernel<<<1, 32, 0, stream>>>(chosen, o_chosen, KITER);
    transpose_W_kernel<<<(int)(((long)N * KITER + 255) / 256), b256, 0, stream>>>(W, o_WT, N, KITER);
    add_mean_kernel<<<(int)((ND + 255) / 256), b256, 0, stream>>>(o_recon, mean, ND, D);
}
